// AttentionSummarization_91018946937106
// MI455X (gfx1250) — compile-verified, matrix operand order checked by reading, not run on hardware
//
#include <hip/hip_runtime.h>
#include <hip/hip_bf16.h>

typedef __attribute__((ext_vector_type(16))) _Float16 v16h;
typedef __attribute__((ext_vector_type(8)))  float    v8f;

#define T_LEN 4096
#define R_REG 49
#define DIM   512
#define NSEG  256
#define NSEG_PAD 258     // row 0 = s=-1 zeros, rows 1..256 = segments, row 257 = s=256 zeros
#define NCLS  106
#define NCLS_PAD 112
#define NCAT  18

// Load a 16-half fragment: halves K = 0..7 from p[0..7], K = 16..23 from p[16..23]
// (caller adds hi = 8*(lane>=16) into p so lanes 16..31 read K+8 per the ISA layout).
__device__ __forceinline__ v16h frag_ld16(const _Float16* __restrict__ p) {
    v16h x;
#pragma unroll
    for (int j = 0; j < 8; ++j) x[j] = p[j];
#pragma unroll
    for (int j = 0; j < 8; ++j) x[8 + j] = p[16 + j];
    return x;
}

// A fragment: 16xK row-major tile, row stride ld. lane: M = lane&15.
__device__ __forceinline__ v16h frag_a_row(const _Float16* __restrict__ P, int ld,
                                           int mbase, int kbase, int lane) {
    int M  = lane & 15;
    int hi = (lane >> 4) << 3;
    return frag_ld16(P + (size_t)(mbase + M) * ld + kbase + hi);
}

// B fragment from K-contiguous storage: Q holds B^T row-major, i.e. element B[K][N] = Q[N*ld + K].
__device__ __forceinline__ v16h frag_b_kcontig(const _Float16* __restrict__ Q, int ld,
                                               int kbase, int nbase, int lane) {
    int N  = lane & 15;
    int hi = (lane >> 4) << 3;
    return frag_ld16(Q + (size_t)(nbase + N) * ld + kbase + hi);
}

// ---------------- kernel 0: pack weights to f16 + zero pad rows of Xsd_pad ----------------
__global__ void pack_weights(const float* __restrict__ W1, const float* __restrict__ W2,
                             _Float16* __restrict__ W1h, _Float16* __restrict__ W2h,
                             _Float16* __restrict__ Xsd_pad) {
    size_t i = (size_t)blockIdx.x * blockDim.x + threadIdx.x;
    const size_t n1 = (size_t)3 * DIM * DIM;               // 786432
    const size_t n2 = (size_t)NCLS_PAD * DIM;              // 57344
    if (i < n1) {
        int k   = (int)(i / ((size_t)DIM * DIM));
        int rem = (int)(i % ((size_t)DIM * DIM));
        int o = rem / DIM, ii = rem % DIM;
        W1h[i] = (_Float16)W1[((size_t)o * DIM + ii) * 3 + k];
    } else if (i < n1 + n2) {
        size_t j = i - n1;
        int m = (int)(j / DIM), ii = (int)(j % DIM);
        W2h[j] = (m < NCLS) ? (_Float16)W2[(size_t)m * DIM + ii] : (_Float16)0.f;
    } else if (i < n1 + n2 + 2 * DIM) {                    // zero rows 0 and 257 of Xsd_pad
        size_t j = i - n1 - n2;                            // 0..1023
        size_t row = (j < DIM) ? 0 : (NSEG_PAD - 1);
        Xsd_pad[row * DIM + (j & (DIM - 1))] = (_Float16)0.f;
    }
}

// ---------------- kernel 1: spatial attention over R=49, one block per t ----------------
__global__ void spatial_attn(const float* __restrict__ feat, const float* __restrict__ att_w,
                             float* __restrict__ fbar, float* __restrict__ asp_out) {
    __shared__ float s_lds[R_REG];
    __shared__ float a_lds[R_REG];
    int t    = blockIdx.x;
    int lane = threadIdx.x & 31;
    int wv   = threadIdx.x >> 5;     // 8 waves
    const float* fbase = feat + (size_t)t * R_REG * DIM;
    const float4* wr4  = (const float4*)att_w;

    for (int r = wv; r < R_REG; r += 8) {
        const float4* fr4 = (const float4*)(fbase + (size_t)r * DIM);
        float acc = 0.f;
#pragma unroll
        for (int j = 0; j < 4; ++j) {
            int idx = lane + 32 * j;           // 128 float4 per row, fully coalesced
            float4 a = fr4[idx], w = wr4[idx];
            acc += a.x * w.x + a.y * w.y + a.z * w.z + a.w * w.w;
        }
#pragma unroll
        for (int off = 16; off > 0; off >>= 1) acc += __shfl_down(acc, off, 32);
        if (lane == 0) s_lds[r] = acc;
    }
    __syncthreads();

    float mx = -1e30f;
    for (int r = 0; r < R_REG; ++r) mx = fmaxf(mx, s_lds[r]);
    float sum = 0.f;
    for (int r = 0; r < R_REG; ++r) sum += expf(s_lds[r] - mx);
    float inv = 1.f / sum;
    if (threadIdx.x < R_REG) {
        float al = expf(s_lds[threadIdx.x] - mx) * inv;
        a_lds[threadIdx.x] = al;
        asp_out[(size_t)t * R_REG + threadIdx.x] = al;
    }
    __syncthreads();

    int d0 = threadIdx.x;                      // each thread: d0 and d0+256
    float acc0 = 0.f, acc1 = 0.f;
    for (int r = 0; r < R_REG; ++r) {
        float al = a_lds[r];
        acc0 += al * fbase[(size_t)r * DIM + d0];
        acc1 += al * fbase[(size_t)r * DIM + d0 + 256];
    }
    fbar[(size_t)t * DIM + d0]       = acc0;
    fbar[(size_t)t * DIM + d0 + 256] = acc1;
}

// ---------------- kernel 2: per-segment mean (seg_ids sorted; binary search) ----------------
__global__ void seg_mean(const int* __restrict__ ids, const float* __restrict__ fbar,
                         _Float16* __restrict__ Xds, _Float16* __restrict__ Xsd_pad) {
    int seg = blockIdx.x;
    int lo = 0, hi = T_LEN;
    while (lo < hi) { int mid = (lo + hi) >> 1; if (ids[mid] < seg) lo = mid + 1; else hi = mid; }
    int a = lo, b = T_LEN;
    while (a < b) { int mid = (a + b) >> 1; if (ids[mid] <= seg) a = mid + 1; else b = mid; }
    int cnt = a - lo;
    float invc = (cnt > 0) ? 1.f / (float)cnt : 0.f;
    for (int d = threadIdx.x; d < DIM; d += blockDim.x) {
        float sum = 0.f;
        for (int t = lo; t < a; ++t) sum += fbar[(size_t)t * DIM + d];
        float mean = sum * invc;
        Xds[(size_t)d * NSEG + seg]            = (_Float16)mean;   // [D,S]   (fk B operand)
        Xsd_pad[(size_t)(seg + 1) * DIM + d]   = (_Float16)mean;   // [S+pad,D] (conv1 B operand)
    }
}

// ---------------- kernel 3: conv1 (kernel=3, SAME) as 3 shifted WMMA GEMMs + bias + ReLU ----------------
// 512 waves: 32 o-tiles x 16 s-tiles; K = 512 per tap. Boundary handled by zero pad rows -> no branches.
__global__ void conv1_wmma(const _Float16* __restrict__ W1h, const _Float16* __restrict__ Xsd_pad,
                           const float* __restrict__ b1, _Float16* __restrict__ Hsd) {
    int wave = blockIdx.x * (blockDim.x >> 5) + (threadIdx.x >> 5);
    int lane = threadIdx.x & 31;
    int mbase = (wave >> 4) * 16;
    int sbase = (wave & 15) * 16;
    int N  = lane & 15;
    int hi = (lane >> 4) << 3;
    v8f c = {};
#pragma unroll
    for (int k = 0; k < 3; ++k) {
        const _Float16* Wk = W1h + (size_t)k * DIM * DIM;
        // B[K=i][N] = X[i][sp] with sp = sbase+N+k-1; padded row index sp+1 in [0,257] always valid
        const _Float16* xrow = Xsd_pad + (size_t)(sbase + N + k) * DIM;
        for (int ib = 0; ib < DIM; ib += 32) {
            v16h av = frag_a_row(Wk, DIM, mbase, ib, lane);
            v16h bv = frag_ld16(xrow + ib + hi);
            c = __builtin_amdgcn_wmma_f32_16x16x32_f16(false, av, false, bv, (short)0, c, false, false);
        }
    }
    int roff = (lane >> 4) << 3;
    // Hsd[s][o]: for fixed lane the 8 outputs are contiguous in o -> one 16B store
    _Float16* hrow = Hsd + (size_t)(sbase + N) * DIM + mbase + roff;
#pragma unroll
    for (int v = 0; v < 8; ++v) {
        int o = mbase + roff + v;
        hrow[v] = (_Float16)fmaxf(c[v] + b1[o], 0.f);
    }
}

// ---------------- kernel 4: conv2 (1x1) -> keysteps [106,256] fp32 ----------------
__global__ void conv2_wmma(const _Float16* __restrict__ W2h, const _Float16* __restrict__ Hsd,
                           const float* __restrict__ b2, float* __restrict__ ks_out) {
    int wave = blockIdx.x * (blockDim.x >> 5) + (threadIdx.x >> 5);   // 112 waves
    int lane = threadIdx.x & 31;
    int mbase = (wave >> 4) * 16;
    int sbase = (wave & 15) * 16;
    v8f c = {};
    for (int kb = 0; kb < DIM; kb += 32) {
        v16h av = frag_a_row(W2h, DIM, mbase, kb, lane);
        v16h bv = frag_b_kcontig(Hsd, DIM, kb, sbase, lane);   // B[K=i][N=s] = Hsd[s*512+i]
        c = __builtin_amdgcn_wmma_f32_16x16x32_f16(false, av, false, bv, (short)0, c, false, false);
    }
    int N = lane & 15, roff = (lane >> 4) << 3;
#pragma unroll
    for (int v = 0; v < 8; ++v) {
        int m = mbase + roff + v;
        if (m < NCLS) ks_out[(size_t)m * NSEG + sbase + N] = c[v] + b2[m];
    }
}

// ---------------- kernel 5: softmax over classes per segment -> f16 attention [112,256] ----------------
__global__ void cls_softmax(const float* __restrict__ ks, _Float16* __restrict__ atth) {
    int s = threadIdx.x;   // 256 threads, one segment each; column reads are coalesced
    float mx = -1e30f;
    for (int m = 0; m < NCLS; ++m) mx = fmaxf(mx, ks[(size_t)m * NSEG + s]);
    float sum = 0.f;
    for (int m = 0; m < NCLS; ++m) sum += expf(ks[(size_t)m * NSEG + s] - mx);
    float inv = 1.f / sum;
    for (int m = 0; m < NCLS_PAD; ++m)
        atth[(size_t)m * NSEG + s] = (m < NCLS) ? (_Float16)(expf(ks[(size_t)m * NSEG + s] - mx) * inv)
                                                : (_Float16)0.f;
}

// ---------------- kernel 6: f_keysteps[m,d] = sum_s att[m,s] * X[d,s]  (WMMA) ----------------
__global__ void fk_wmma(const _Float16* __restrict__ atth, const _Float16* __restrict__ Xds,
                        float* __restrict__ FK) {
    int wave = blockIdx.x * (blockDim.x >> 5) + (threadIdx.x >> 5);   // 224 waves: 7 m-tiles x 32 d-tiles
    int lane = threadIdx.x & 31;
    int mbase = (wave >> 5) * 16;
    int dbase = (wave & 31) * 16;
    v8f c = {};
    for (int kb = 0; kb < NSEG; kb += 32) {
        v16h av = frag_a_row(atth, NSEG, mbase, kb, lane);
        v16h bv = frag_b_kcontig(Xds, NSEG, kb, dbase, lane);  // B[K=s][N=d] = Xds[d*256+s]
        c = __builtin_amdgcn_wmma_f32_16x16x32_f16(false, av, false, bv, (short)0, c, false, false);
    }
    int N = lane & 15, roff = (lane >> 4) << 3;
#pragma unroll
    for (int v = 0; v < 8; ++v) {
        int m = mbase + roff + v;
        if (m < NCLS) FK[(size_t)m * DIM + dbase + N] = c[v];
    }
}

// ---------------- kernel 7: keystep attention + category head ----------------
__global__ void finalize(const float* __restrict__ FK, const float* __restrict__ att2_w,
                         const float* __restrict__ cls_w, const float* __restrict__ cls_b,
                         float* __restrict__ out_cats, float* __restrict__ out_aks) {
    __shared__ float s2[NCLS];
    __shared__ float a2[NCLS];
    __shared__ float fv[DIM];
    int lane = threadIdx.x & 31;
    int wv   = threadIdx.x >> 5;   // 4 waves
    for (int m = wv; m < NCLS; m += 4) {
        const float* row = FK + (size_t)m * DIM;
        float acc = 0.f;
#pragma unroll
        for (int j = 0; j < 16; ++j) { int d = lane + 32 * j; acc += row[d] * att2_w[d]; }
#pragma unroll
        for (int off = 16; off > 0; off >>= 1) acc += __shfl_down(acc, off, 32);
        if (lane == 0) s2[m] = acc;
    }
    __syncthreads();
    float mx = -1e30f;
    for (int m = 0; m < NCLS; ++m) mx = fmaxf(mx, s2[m]);
    float sum = 0.f;
    for (int m = 0; m < NCLS; ++m) sum += expf(s2[m] - mx);
    float inv = 1.f / sum;
    if (threadIdx.x < NCLS) {
        float al = expf(s2[threadIdx.x] - mx) * inv;
        a2[threadIdx.x] = al;
        out_aks[threadIdx.x] = al;
    }
    __syncthreads();
    for (int d = threadIdx.x; d < DIM; d += 128) {
        float acc = 0.f;
        for (int m = 0; m < NCLS; ++m) acc += FK[(size_t)m * DIM + d] * a2[m];
        fv[d] = acc;
    }
    __syncthreads();
    if (threadIdx.x < NCAT) {
        int cix = threadIdx.x;
        float acc = cls_b[cix];
        for (int d = 0; d < DIM; ++d) acc += fv[d] * cls_w[(size_t)d * NCAT + cix];
        out_cats[cix] = acc;
    }
}

extern "C" void kernel_launch(void* const* d_in, const int* in_sizes, int n_in,
                              void* d_out, int out_size, void* d_ws, size_t ws_size,
                              hipStream_t stream) {
    const float* feature = (const float*)d_in[0];
    const int*   seg_ids = (const int*)d_in[1];
    const float* att_w   = (const float*)d_in[2];
    // d_in[3] att_b: constant over softmax axis -> no effect
    const float* att2_w  = (const float*)d_in[4];
    // d_in[5] att2_b: constant over softmax axis -> no effect
    const float* W1      = (const float*)d_in[6];
    const float* b1      = (const float*)d_in[7];
    const float* W2      = (const float*)d_in[8];
    const float* b2      = (const float*)d_in[9];
    const float* cls_w   = (const float*)d_in[10];
    const float* cls_b   = (const float*)d_in[11];

    float* out      = (float*)d_out;
    float* out_ks   = out;               // [106,256]  = 27136
    float* out_cats = out + 27136;       // [18]
    float* out_asp  = out + 27154;       // [4096,49]  = 200704
    float* out_aks  = out + 227858;      // [106]

    char* ws = (char*)d_ws;
    float*    fbar    = (float*)ws;    ws += sizeof(float) * (size_t)T_LEN * DIM;       // 8 MB
    float*    FK      = (float*)ws;    ws += sizeof(float) * (size_t)NCLS * DIM;
    _Float16* W1h     = (_Float16*)ws; ws += sizeof(_Float16) * (size_t)3 * DIM * DIM;
    _Float16* W2h     = (_Float16*)ws; ws += sizeof(_Float16) * (size_t)NCLS_PAD * DIM;
    _Float16* Xds     = (_Float16*)ws; ws += sizeof(_Float16) * (size_t)DIM * NSEG;
    _Float16* Xsd_pad = (_Float16*)ws; ws += sizeof(_Float16) * (size_t)NSEG_PAD * DIM;
    _Float16* Hsd     = (_Float16*)ws; ws += sizeof(_Float16) * (size_t)NSEG * DIM;
    _Float16* atth    = (_Float16*)ws; ws += sizeof(_Float16) * (size_t)NCLS_PAD * NSEG;

    pack_weights<<<3300, 256, 0, stream>>>(W1, W2, W1h, W2h, Xsd_pad);  // 844800 threads exactly
    spatial_attn<<<T_LEN, 256, 0, stream>>>(feature, att_w, fbar, out_asp);
    seg_mean<<<NSEG, 256, 0, stream>>>(seg_ids, fbar, Xds, Xsd_pad);
    conv1_wmma<<<128, 128, 0, stream>>>(W1h, Xsd_pad, b1, Hsd);         // 512 waves
    conv2_wmma<<<28, 128, 0, stream>>>(W2h, Hsd, b2, out_ks);           // 112 waves
    cls_softmax<<<1, 256, 0, stream>>>(out_ks, atth);
    fk_wmma<<<56, 128, 0, stream>>>(atth, Xds, FK);                     // 224 waves
    finalize<<<1, 128, 0, stream>>>(FK, att2_w, cls_w, cls_b, out_cats, out_aks);
}
